// MoE_58772332479041
// MI455X (gfx1250) — compile-verified, matrix operand order checked by reading, not run on hardware
//
#include <hip/hip_runtime.h>
#include <hip/hip_bf16.h>

#define NUM_EXPERTS 8
#define TOPK 2
#define DIM 2048
#define HIDDEN 2048
#define NTOK 16384                    // B*S = 4*4096
#define NROWS (NTOK * TOPK)           // 32768 routed rows
#define TM 128
#define TN 128
#define TK 32
#define MAX_TILES (NROWS / TM + NUM_EXPERTS)   // 264

typedef __attribute__((ext_vector_type(16))) __bf16 v16bf;
typedef __attribute__((ext_vector_type(8)))  __bf16 v8bf;
typedef __attribute__((ext_vector_type(8)))  float  v8f;
typedef __attribute__((ext_vector_type(4)))  unsigned int v4u;

struct MoeMeta {
  int   counts[NUM_EXPERTS];
  int   offsets[NUM_EXPERTS];
  float sum_scores[NUM_EXPERTS];
  float rz_accum;
  int   total_tiles;
  int   tile_row0[MAX_TILES];
  int   tile_rowend[MAX_TILES];
  int   tile_expert[MAX_TILES];
};

__device__ __forceinline__ unsigned int bf16_of(float f) {
  unsigned u = __builtin_bit_cast(unsigned, f);
  u += 0x7FFFu + ((u >> 16) & 1u);    // round-to-nearest-even
  return u >> 16;
}
__device__ __forceinline__ unsigned int pack2(float a, float b) {
  return bf16_of(a) | (bf16_of(b) << 16);
}

// Load a 16-element bf16 WMMA operand from LDS: two contiguous 16B chunks
// (K = kb..kb+7 and K = kb+16..kb+23 in the CDNA5 16-bit A/B layout).
__device__ __forceinline__ v16bf frag16(const unsigned short* p, int kb) {
  v8bf lo = *(const v8bf*)(p + kb);
  v8bf hi = *(const v8bf*)(p + kb + 16);
  return __builtin_shufflevector(lo, hi, 0, 1, 2, 3, 4, 5, 6, 7,
                                 8, 9, 10, 11, 12, 13, 14, 15);
}

// ---------------------------------------------------------------- init
__global__ void init_meta_kernel(MoeMeta* meta) {
  if (threadIdx.x == 0 && blockIdx.x == 0) {
    for (int e = 0; e < NUM_EXPERTS; ++e) { meta->counts[e] = 0; meta->sum_scores[e] = 0.f; }
    meta->rz_accum = 0.f;
    meta->total_tiles = 0;
  }
}

__global__ __launch_bounds__(256) void zero_out_kernel(float4* out) {
  size_t i = (size_t)blockIdx.x * blockDim.x + threadIdx.x;   // NTOK*DIM/4 elements
  out[i] = make_float4(0.f, 0.f, 0.f, 0.f);
}

// ---------------------------------------------------------------- router
__global__ __launch_bounds__(256) void router_kernel(
    const float* __restrict__ x, const float* __restrict__ wr,
    int* __restrict__ pair_e, int* __restrict__ pair_r, float* __restrict__ pair_s,
    MoeMeta* meta) {
  __shared__ float red[NUM_EXPERTS * 256];
  const int n   = blockIdx.x;
  const int tid = threadIdx.x;
  const float* xr = x + (size_t)n * DIM;

  float acc[NUM_EXPERTS];
#pragma unroll
  for (int e = 0; e < NUM_EXPERTS; ++e) acc[e] = 0.f;
  for (int d = tid; d < DIM; d += 256) {
    float xv = xr[d];
    const float* w = wr + d * NUM_EXPERTS;
#pragma unroll
    for (int e = 0; e < NUM_EXPERTS; ++e) acc[e] += xv * w[e];
  }
#pragma unroll
  for (int e = 0; e < NUM_EXPERTS; ++e) red[e * 256 + tid] = acc[e];
  __syncthreads();
  for (int s = 128; s > 0; s >>= 1) {
    if (tid < s) {
#pragma unroll
      for (int e = 0; e < NUM_EXPERTS; ++e) red[e * 256 + tid] += red[e * 256 + tid + s];
    }
    __syncthreads();
  }
  if (tid == 0) {
    float logit[NUM_EXPERTS], m = -1e30f;
#pragma unroll
    for (int e = 0; e < NUM_EXPERTS; ++e) { logit[e] = red[e * 256]; m = fmaxf(m, logit[e]); }
    float se = 0.f;
#pragma unroll
    for (int e = 0; e < NUM_EXPERTS; ++e) se += __expf(logit[e] - m);
    float lse = m + __logf(se);
    atomicAdd(&meta->rz_accum, lse * lse);
    float score[NUM_EXPERTS];
    float inv = 1.f / se;
#pragma unroll
    for (int e = 0; e < NUM_EXPERTS; ++e) score[e] = __expf(logit[e] - m) * inv;
    int i1 = 0;
#pragma unroll
    for (int e = 1; e < NUM_EXPERTS; ++e) if (score[e] > score[i1]) i1 = e;
    int i2 = (i1 == 0) ? 1 : 0;
#pragma unroll
    for (int e = 0; e < NUM_EXPERTS; ++e)
      if (e != i1 && score[e] > score[i2]) i2 = e;
    int idxs[2] = { i1, i2 };
#pragma unroll
    for (int k = 0; k < TOPK; ++k) {
      int e = idxs[k]; float sc = score[e];
      int r = atomicAdd(&meta->counts[e], 1);
      atomicAdd(&meta->sum_scores[e], sc);
      pair_e[n * TOPK + k] = e;
      pair_r[n * TOPK + k] = r;
      pair_s[n * TOPK + k] = sc;
    }
  }
}

// ---------------------------------------------------------------- offsets + tile map
__global__ void tilemap_kernel(MoeMeta* meta) {
  if (threadIdx.x != 0 || blockIdx.x != 0) return;
  int off = 0, t = 0;
  for (int e = 0; e < NUM_EXPERTS; ++e) {
    meta->offsets[e] = off;
    int c = meta->counts[e];
    int nt = (c + TM - 1) / TM;
    for (int i = 0; i < nt; ++i) {
      meta->tile_row0[t]   = off + i * TM;
      meta->tile_rowend[t] = off + c;
      meta->tile_expert[t] = e;
      ++t;
    }
    off += c;
  }
  meta->total_tiles = t;
}

// ---------------------------------------------------------------- scatter
__global__ __launch_bounds__(256) void scatter_kernel(
    const int* __restrict__ pair_e, const int* __restrict__ pair_r,
    const float* __restrict__ pair_s, const MoeMeta* __restrict__ meta,
    int* __restrict__ row_token, float* __restrict__ row_score) {
  int p = blockIdx.x * 256 + threadIdx.x;
  if (p >= NROWS) return;
  int e = pair_e[p];
  int t = meta->offsets[e] + pair_r[p];
  row_token[t] = p >> 1;    // TOPK == 2
  row_score[t] = pair_s[p];
}

// ---------------------------------------------------------------- fused w1/w3 GEMM + SwiGLU
// LDS layouts: sA row-major [row][TK]; sB1/sB3 COLUMN-major [n][TK] so every
// lane's WMMA fragment is two contiguous 16B runs (ds_load_b128 x2).
__global__ __launch_bounds__(256) void gemm13_kernel(
    const float* __restrict__ x, const float* __restrict__ w1, const float* __restrict__ w3,
    const int* __restrict__ row_token, const float* __restrict__ row_score,
    const MoeMeta* __restrict__ meta, unsigned short* __restrict__ hbuf) {
  const int tile = blockIdx.y;
  if (tile >= meta->total_tiles) return;
  const int n0      = blockIdx.x * TN;
  const int row0    = meta->tile_row0[tile];
  const int row_end = meta->tile_rowend[tile];
  const int e       = meta->tile_expert[tile];
  const float* W1 = w1 + (size_t)e * DIM * HIDDEN;
  const float* W3 = w3 + (size_t)e * DIM * HIDDEN;

  __shared__ __align__(16) unsigned short sA[TM * TK];
  __shared__ __align__(16) unsigned short sB1[TN * TK];
  __shared__ __align__(16) unsigned short sB3[TN * TK];

  const int tid  = threadIdx.x;
  const int lane = tid & 31;
  const int wave = tid >> 5;
  const int wm   = wave & 3;      // 4 wave-rows of 32
  const int wn   = wave >> 2;     // 2 wave-cols of 64

  v8f acc1[2][4] = {};
  v8f acc3[2][4] = {};

  // A staging: thread -> (row ar, 16-col chunk ac); gather row invariant over k
  const int ar = tid >> 1;
  const int ac = (tid & 1) * 16;
  const int grow = row0 + ar;
  const float* xrow = nullptr; float scale = 0.f;
  if (grow < row_end) { xrow = x + (size_t)row_token[grow] * DIM; scale = row_score[grow]; }

  // B staging: thread -> (column bn, 16-row K chunk bk)
  const int bn = tid & 127;
  const int bk = (tid >> 7) * 16;

  const int lrow = lane & 15;
  const int kb   = (lane >> 4) * 8;

  for (int k0 = 0; k0 < DIM; k0 += TK) {
    __syncthreads();
    // ---- stage A (row-major, packed bf16 pairs, two b128 stores)
    {
      v4u pk0 = {0, 0, 0, 0}, pk1 = {0, 0, 0, 0};
      if (xrow) {
        const float4* xs = (const float4*)(xrow + k0 + ac);
        float4 f0 = xs[0], f1 = xs[1], f2 = xs[2], f3 = xs[3];
        pk0 = (v4u){ pack2(f0.x * scale, f0.y * scale), pack2(f0.z * scale, f0.w * scale),
                     pack2(f1.x * scale, f1.y * scale), pack2(f1.z * scale, f1.w * scale) };
        pk1 = (v4u){ pack2(f2.x * scale, f2.y * scale), pack2(f2.z * scale, f2.w * scale),
                     pack2(f3.x * scale, f3.y * scale), pack2(f3.z * scale, f3.w * scale) };
      }
      v4u* d = (v4u*)&sA[ar * TK + ac];
      d[0] = pk0; d[1] = pk1;
    }
    // ---- stage B1/B3 (column-major: one column per thread, 16 K values)
    {
      const float* p1 = W1 + (size_t)(k0 + bk) * HIDDEN + n0 + bn;
      const float* p3 = W3 + (size_t)(k0 + bk) * HIDDEN + n0 + bn;
      unsigned int q1[8], q3[8];
#pragma unroll
      for (int j = 0; j < 8; ++j) {
        q1[j] = pack2(p1[(2 * j) * HIDDEN], p1[(2 * j + 1) * HIDDEN]);
        q3[j] = pack2(p3[(2 * j) * HIDDEN], p3[(2 * j + 1) * HIDDEN]);
      }
      v4u* d1 = (v4u*)&sB1[bn * TK + bk];
      v4u* d3 = (v4u*)&sB3[bn * TK + bk];
      d1[0] = (v4u){q1[0], q1[1], q1[2], q1[3]};
      d1[1] = (v4u){q1[4], q1[5], q1[6], q1[7]};
      d3[0] = (v4u){q3[0], q3[1], q3[2], q3[3]};
      d3[1] = (v4u){q3[4], q3[5], q3[6], q3[7]};
      if (k0 + TK < DIM) {
        __builtin_prefetch(p1 + (size_t)TK * HIDDEN, 0, 0);
        __builtin_prefetch(p3 + (size_t)TK * HIDDEN, 0, 0);
      }
    }
    __syncthreads();

    v16bf afrag[2];
#pragma unroll
    for (int mi = 0; mi < 2; ++mi)
      afrag[mi] = frag16(&sA[(wm * 32 + mi * 16 + lrow) * TK], kb);
#pragma unroll
    for (int ni = 0; ni < 4; ++ni) {
      const int n = wn * 64 + ni * 16 + lrow;
      v16bf b1 = frag16(&sB1[n * TK], kb);
      v16bf b3 = frag16(&sB3[n * TK], kb);
#pragma unroll
      for (int mi = 0; mi < 2; ++mi) {
        acc1[mi][ni] = __builtin_amdgcn_wmma_f32_16x16x32_bf16(
            false, afrag[mi], false, b1, (short)0, acc1[mi][ni], false, false);
        acc3[mi][ni] = __builtin_amdgcn_wmma_f32_16x16x32_bf16(
            false, afrag[mi], false, b3, (short)0, acc3[mi][ni], false, false);
      }
    }
  }

  // SwiGLU epilogue -> h (bf16)
  const int rhi = (lane >> 4) * 8;
#pragma unroll
  for (int mi = 0; mi < 2; ++mi) {
#pragma unroll
    for (int ni = 0; ni < 4; ++ni) {
      const int colg = n0 + wn * 64 + ni * 16 + lrow;
#pragma unroll
      for (int v = 0; v < 8; ++v) {
        const int rg = row0 + wm * 32 + mi * 16 + v + rhi;
        if (rg < row_end) {
          float a = acc1[mi][ni][v];
          float c = acc3[mi][ni][v];
          float hv = (a / (1.f + __expf(-a))) * c;    // silu(a)*c
          hbuf[(size_t)rg * HIDDEN + colg] = (unsigned short)bf16_of(hv);
        }
      }
    }
  }
}

// ---------------------------------------------------------------- w2 GEMM + scatter-add combine
__global__ __launch_bounds__(256) void gemm2_kernel(
    const unsigned short* __restrict__ hbuf, const float* __restrict__ w2,
    const int* __restrict__ row_token, const MoeMeta* __restrict__ meta,
    float* __restrict__ out) {
  const int tile = blockIdx.y;
  if (tile >= meta->total_tiles) return;
  const int n0      = blockIdx.x * TN;
  const int row0    = meta->tile_row0[tile];
  const int row_end = meta->tile_rowend[tile];
  const int e       = meta->tile_expert[tile];
  const float* W2 = w2 + (size_t)e * HIDDEN * DIM;

  __shared__ __align__(16) unsigned short sA[TM * TK];
  __shared__ __align__(16) unsigned short sB[TN * TK];   // column-major [n][TK]

  const int tid  = threadIdx.x;
  const int lane = tid & 31;
  const int wave = tid >> 5;
  const int wm   = wave & 3;
  const int wn   = wave >> 2;

  v8f acc[2][4] = {};

  const int ar = tid >> 1;
  const int ac = (tid & 1) * 16;
  const int grow = row0 + ar;
  const bool arow_ok = (grow < row_end);
  const unsigned short* hrow = hbuf + (size_t)grow * HIDDEN;

  const int bn = tid & 127;
  const int bk = (tid >> 7) * 16;
  const int lrow = lane & 15;
  const int kb   = (lane >> 4) * 8;

  for (int k0 = 0; k0 < HIDDEN; k0 += TK) {
    __syncthreads();
    // ---- stage A (already bf16: straight 16B copies)
    {
      v4u a0 = {0, 0, 0, 0}, a1 = {0, 0, 0, 0};
      if (arow_ok) {
        const v4u* hs = (const v4u*)(hrow + k0 + ac);
        a0 = hs[0]; a1 = hs[1];
      }
      v4u* d = (v4u*)&sA[ar * TK + ac];
      d[0] = a0; d[1] = a1;
    }
    // ---- stage B (column-major)
    {
      const float* bp = W2 + (size_t)(k0 + bk) * DIM + n0 + bn;
      unsigned int q[8];
#pragma unroll
      for (int j = 0; j < 8; ++j)
        q[j] = pack2(bp[(2 * j) * DIM], bp[(2 * j + 1) * DIM]);
      v4u* d = (v4u*)&sB[bn * TK + bk];
      d[0] = (v4u){q[0], q[1], q[2], q[3]};
      d[1] = (v4u){q[4], q[5], q[6], q[7]};
      if (k0 + TK < HIDDEN) __builtin_prefetch(bp + (size_t)TK * DIM, 0, 0);
    }
    __syncthreads();

    v16bf afrag[2];
#pragma unroll
    for (int mi = 0; mi < 2; ++mi)
      afrag[mi] = frag16(&sA[(wm * 32 + mi * 16 + lrow) * TK], kb);
#pragma unroll
    for (int ni = 0; ni < 4; ++ni) {
      const int n = wn * 64 + ni * 16 + lrow;
      v16bf bfrag = frag16(&sB[n * TK], kb);
#pragma unroll
      for (int mi = 0; mi < 2; ++mi)
        acc[mi][ni] = __builtin_amdgcn_wmma_f32_16x16x32_bf16(
            false, afrag[mi], false, bfrag, (short)0, acc[mi][ni], false, false);
    }
  }

  const int rhi = (lane >> 4) * 8;
#pragma unroll
  for (int mi = 0; mi < 2; ++mi) {
#pragma unroll
    for (int ni = 0; ni < 4; ++ni) {
      const int colg = n0 + wn * 64 + ni * 16 + lrow;
#pragma unroll
      for (int v = 0; v < 8; ++v) {
        const int rg = row0 + wm * 32 + mi * 16 + v + rhi;
        if (rg < row_end) {
          int token = row_token[rg];
          atomicAdd(&out[(size_t)token * DIM + colg], acc[mi][ni][v]);
        }
      }
    }
  }
}

// ---------------------------------------------------------------- aux losses
__global__ void loss_kernel(const MoeMeta* meta, float* out_tail) {
  if (threadIdx.x != 0 || blockIdx.x != 0) return;
  float lb = 0.f;
  for (int e = 0; e < NUM_EXPERTS; ++e) {
    float c   = (float)meta->counts[e];
    float avg = meta->sum_scores[e] / fmaxf(c, 1.f);
    lb += (c / (float)NROWS) * avg;
  }
  out_tail[0] = lb * (float)NUM_EXPERTS;
  out_tail[1] = meta->rz_accum / (float)NTOK;
}

// ---------------------------------------------------------------- launch
extern "C" void kernel_launch(void* const* d_in, const int* in_sizes, int n_in,
                              void* d_out, int out_size, void* d_ws, size_t ws_size,
                              hipStream_t stream) {
  const float* x  = (const float*)d_in[0];
  const float* wr = (const float*)d_in[1];
  const float* w1 = (const float*)d_in[2];
  const float* w2 = (const float*)d_in[3];
  const float* w3 = (const float*)d_in[4];
  float* out = (float*)d_out;

  char* ws = (char*)d_ws;
  size_t o = 0;
  unsigned short* hbuf = (unsigned short*)(ws + o); o += (size_t)NROWS * HIDDEN * sizeof(unsigned short);
  int*   row_token = (int*)(ws + o);   o += (size_t)NROWS * sizeof(int);
  float* row_score = (float*)(ws + o); o += (size_t)NROWS * sizeof(float);
  int*   pair_e = (int*)(ws + o);      o += (size_t)NROWS * sizeof(int);
  int*   pair_r = (int*)(ws + o);      o += (size_t)NROWS * sizeof(int);
  float* pair_s = (float*)(ws + o);    o += (size_t)NROWS * sizeof(float);
  MoeMeta* meta = (MoeMeta*)(ws + o);

  init_meta_kernel<<<1, 32, 0, stream>>>(meta);
  router_kernel<<<NTOK, 256, 0, stream>>>(x, wr, pair_e, pair_r, pair_s, meta);
  tilemap_kernel<<<1, 1, 0, stream>>>(meta);
  scatter_kernel<<<NROWS / 256, 256, 0, stream>>>(pair_e, pair_r, pair_s, meta,
                                                  row_token, row_score);
  zero_out_kernel<<<((size_t)NTOK * DIM / 4) / 256, 256, 0, stream>>>((float4*)out);

  dim3 g13(HIDDEN / TN, MAX_TILES);
  gemm13_kernel<<<g13, 256, 0, stream>>>(x, w1, w3, row_token, row_score, meta, hbuf);

  dim3 g2(DIM / TN, MAX_TILES);
  gemm2_kernel<<<g2, 256, 0, stream>>>(hbuf, w2, row_token, meta, out);

  loss_kernel<<<1, 1, 0, stream>>>(meta, out + (size_t)NTOK * DIM);
}
